// PodDeepONet_42245298323680
// MI455X (gfx1250) — matrix operand order repaired and checked by reading, not compiled, not run on hardware
//
#include <hip/hip_runtime.h>
#include <hip/hip_bf16.h>
#include <math.h>

#define B_ROWS  4096
#define N_COLS  8192
#define P_DIM   256
#define NOISE_F 128
#define WIDTH   32

typedef _Float16 h8_t  __attribute__((ext_vector_type(8)));
typedef _Float16 v16h  __attribute__((ext_vector_type(16)));
typedef float    v8f   __attribute__((ext_vector_type(8)));

// ---------------------------------------------------------------------------
// Kernel 1: branch MLP  128 ->32->32->32->32-> 256 (tanh hidden layers).
// One wave per row; each lane owns one hidden unit (WIDTH == wave32 == 32).
// ---------------------------------------------------------------------------
__global__ __launch_bounds__(256) void pod_branch_mlp(
    const float* __restrict__ noise,
    const float* __restrict__ W1, const float* __restrict__ b1,
    const float* __restrict__ W2, const float* __restrict__ b2,
    const float* __restrict__ W3, const float* __restrict__ b3,
    const float* __restrict__ W4, const float* __restrict__ b4,
    const float* __restrict__ W5, const float* __restrict__ b5,
    _Float16* __restrict__ out_b)
{
    __shared__ float sn[8][NOISE_F];
    const int lane = threadIdx.x;          // 0..31
    const int wv   = threadIdx.y;          // 0..7
    const int row  = blockIdx.x * 8 + wv;

    const float4* src = (const float4*)(noise + (size_t)row * NOISE_F);
    ((float4*)sn[wv])[lane] = src[lane];
    __syncthreads();

    float acc = b1[lane];
    #pragma unroll 4
    for (int k = 0; k < NOISE_F; ++k)
        acc = fmaf(sn[wv][k], W1[k * WIDTH + lane], acc);
    float h = tanhf(acc);

    const float* Ws[3] = { W2, W3, W4 };
    const float* bs[3] = { b2, b3, b4 };
    for (int l = 0; l < 3; ++l) {
        float a2 = bs[l][lane];
        #pragma unroll
        for (int j = 0; j < WIDTH; ++j)
            a2 = fmaf(__shfl(h, j, 32), Ws[l][j * WIDTH + lane], a2);
        h = tanhf(a2);
    }

    float o[8];
    #pragma unroll
    for (int u = 0; u < 8; ++u) o[u] = b5[u * 32 + lane];
    #pragma unroll
    for (int j = 0; j < WIDTH; ++j) {
        const float hj = __shfl(h, j, 32);
        #pragma unroll
        for (int u = 0; u < 8; ++u)
            o[u] = fmaf(hj, W5[j * P_DIM + u * 32 + lane], o[u]);
    }
    #pragma unroll
    for (int u = 0; u < 8; ++u)
        out_b[(size_t)row * P_DIM + u * 32 + lane] = (_Float16)o[u];
}

// ---------------------------------------------------------------------------
// Kernel 2: trunk affine maps.
// ---------------------------------------------------------------------------
__global__ __launch_bounds__(256) void pod_trunk(
    const float* __restrict__ coords,
    const float* __restrict__ Wc, const float* __restrict__ bc,
    const float* __restrict__ Wm, const float* __restrict__ bm,
    _Float16* __restrict__ out_t, float* __restrict__ out_bias)
{
    const int n = blockIdx.x;
    const int p = threadIdx.x;
    const float c0 = coords[n * 2 + 0];
    const float c1 = coords[n * 2 + 1];
    const float v = fmaf(c0, Wc[p], fmaf(c1, Wc[P_DIM + p], bc[p]));
    out_t[(size_t)n * P_DIM + p] = (_Float16)v;
    if (p == 0)
        out_bias[n] = fmaf(c0, Wm[0], fmaf(c1, Wm[1], bm[0]));
}

// ---------------------------------------------------------------------------
// Kernel 3: g = out_b @ out_t^T + bias via v_wmma_f32_16x16x32_f16.
// 8-wave block computes 256x128; LDS-staged, double-buffered K slabs.
// Each wave owns a 64x64 quadrant (4x4 tiles of 16x16, f32 accumulate).
// ---------------------------------------------------------------------------
__global__ __launch_bounds__(256, 1) void pod_gemm_wmma(
    const _Float16* __restrict__ Af,   // out_b: 4096 x 256 f16
    const _Float16* __restrict__ Bf,   // out_t: 8192 x 256 f16
    const float*    __restrict__ bias, // 8192 f32
    float*          __restrict__ out)  // 4096 x 8192 f32
{
    constexpr int LDK      = 40;              // padded slab row stride (halves)
    constexpr int A_HALVES = 256 * LDK;       // 10240
    constexpr int B_HALVES = 128 * LDK;       // 5120
    __shared__ _Float16 lds[2 * A_HALVES + 2 * B_HALVES];   // 60 KB

    const int lane  = threadIdx.x;
    const int wv    = threadIdx.y;            // 0..7
    const int t     = wv * 32 + lane;         // 0..255
    const int wm    = wv >> 1;                // 0..3 (M quadrant)
    const int wn    = wv & 1;                 // 0..1 (N quadrant)
    const int bm    = blockIdx.x >> 6;        // 0..15
    const int bn    = blockIdx.x & 63;        // 0..63
    const int mBase = bm * 256;
    const int nBase = bn * 128;
    const int l15   = lane & 15;
    const int hiH   = lane >> 4;
    const int aoff  = hiH ? 8  : 0;           // A K-chunk select (ISA layout)
    const int boff  = hiH ? 8  : 0;           // B K-chunk select (ISA layout)

    // staging decomposition: slab chunk -> (row, 8-half column group)
    const int arow0 = t >> 2, acc0 = (t & 3) * 8;

    v8f acc[4][4];
    #pragma unroll
    for (int i = 0; i < 4; ++i)
        #pragma unroll
        for (int j = 0; j < 4; ++j)
            acc[i][j] = v8f{};

    h8_t areg[4], breg[2];

    // prologue: slab 0
    #pragma unroll
    for (int q = 0; q < 4; ++q)
        areg[q] = *(const h8_t*)(Af + (size_t)(mBase + arow0 + 64 * q) * P_DIM + acc0);
    #pragma unroll
    for (int q = 0; q < 2; ++q)
        breg[q] = *(const h8_t*)(Bf + (size_t)(nBase + arow0 + 64 * q) * P_DIM + acc0);
    #pragma unroll
    for (int q = 0; q < 4; ++q)
        *(h8_t*)(lds + (arow0 + 64 * q) * LDK + acc0) = areg[q];
    #pragma unroll
    for (int q = 0; q < 2; ++q)
        *(h8_t*)(lds + 2 * A_HALVES + (arow0 + 64 * q) * LDK + acc0) = breg[q];
    __syncthreads();

    for (int kb = 0; kb < 8; ++kb) {
        const int cur = kb & 1;
        _Float16* const Acur = lds + cur * A_HALVES;
        _Float16* const Bcur = lds + 2 * A_HALVES + cur * B_HALVES;

        // issue next slab's global loads before consuming LDS
        if (kb < 7) {
            const int k0 = (kb + 1) * 32;
            #pragma unroll
            for (int q = 0; q < 4; ++q)
                areg[q] = *(const h8_t*)(Af + (size_t)(mBase + arow0 + 64 * q) * P_DIM + k0 + acc0);
            #pragma unroll
            for (int q = 0; q < 2; ++q)
                breg[q] = *(const h8_t*)(Bf + (size_t)(nBase + arow0 + 64 * q) * P_DIM + k0 + acc0);
        }

        // fragments from LDS (conflict-free: 80B row stride = 20-bank skew)
        v16h a[4], b[4];
        #pragma unroll
        for (int i = 0; i < 4; ++i) {
            const _Float16* pa = Acur + (wm * 64 + i * 16 + l15) * LDK + aoff;
            const h8_t lo = *(const h8_t*)pa;
            const h8_t hi = *(const h8_t*)(pa + 16);
            #pragma unroll
            for (int e = 0; e < 8; ++e) { a[i][e] = lo[e]; a[i][e + 8] = hi[e]; }
        }
        #pragma unroll
        for (int j = 0; j < 4; ++j) {
            const _Float16* pb = Bcur + (wn * 64 + j * 16 + l15) * LDK + boff;
            const h8_t lo = *(const h8_t*)pb;
            const h8_t hi = *(const h8_t*)(pb + 16);
            #pragma unroll
            for (int e = 0; e < 8; ++e) { b[j][e] = lo[e]; b[j][e + 8] = hi[e]; }
        }

        #pragma unroll
        for (int i = 0; i < 4; ++i)
            #pragma unroll
            for (int j = 0; j < 4; ++j)
                acc[i][j] = __builtin_amdgcn_wmma_f32_16x16x32_f16(
                    false, a[i], false, b[j], (short)0, acc[i][j], false, false);

        // stage next slab into the other buffer, concurrent with compute tail
        if (kb < 7) {
            const int nxt = cur ^ 1;
            _Float16* const Anxt = lds + nxt * A_HALVES;
            _Float16* const Bnxt = lds + 2 * A_HALVES + nxt * B_HALVES;
            #pragma unroll
            for (int q = 0; q < 4; ++q)
                *(h8_t*)(Anxt + (arow0 + 64 * q) * LDK + acc0) = areg[q];
            #pragma unroll
            for (int q = 0; q < 2; ++q)
                *(h8_t*)(Bnxt + (arow0 + 64 * q) * LDK + acc0) = breg[q];
        }
        __syncthreads();
    }

    // epilogue: lane holds col n = 16*tile + l15; VGPR r holds row r + 8*hiH
    const int m0 = mBase + wm * 64;
    const int n0 = nBase + wn * 64;
    #pragma unroll
    for (int j = 0; j < 4; ++j) {
        const int n  = n0 + j * 16 + l15;
        const float bv = bias[n];
        #pragma unroll
        for (int i = 0; i < 4; ++i) {
            const int rbase = m0 + i * 16 + hiH * 8;
            #pragma unroll
            for (int r = 0; r < 8; ++r)
                out[(size_t)(rbase + r) * N_COLS + n] = acc[i][j][r] + bv;
        }
    }
}

// ---------------------------------------------------------------------------
extern "C" void kernel_launch(void* const* d_in, const int* in_sizes, int n_in,
                              void* d_out, int out_size, void* d_ws, size_t ws_size,
                              hipStream_t stream) {
    const float* noise  = (const float*)d_in[0];
    const float* coords = (const float*)d_in[1];
    const float* W1 = (const float*)d_in[2];  const float* b1 = (const float*)d_in[3];
    const float* W2 = (const float*)d_in[4];  const float* b2 = (const float*)d_in[5];
    const float* W3 = (const float*)d_in[6];  const float* b3 = (const float*)d_in[7];
    const float* W4 = (const float*)d_in[8];  const float* b4 = (const float*)d_in[9];
    const float* W5 = (const float*)d_in[10]; const float* b5 = (const float*)d_in[11];
    const float* Wc = (const float*)d_in[12]; const float* bc = (const float*)d_in[13];
    const float* Wm = (const float*)d_in[14]; const float* bm = (const float*)d_in[15];

    char* ws = (char*)d_ws;
    _Float16* out_b    = (_Float16*)ws;                                  // 2 MB
    _Float16* out_t    = (_Float16*)(ws + (size_t)B_ROWS * P_DIM * 2);   // 4 MB
    float*    out_bias = (float*)(ws + (size_t)B_ROWS * P_DIM * 2
                                      + (size_t)N_COLS * P_DIM * 2);     // 32 KB
    float* out = (float*)d_out;

    pod_branch_mlp<<<B_ROWS / 8, dim3(32, 8), 0, stream>>>(
        noise, W1, b1, W2, b2, W3, b3, W4, b4, W5, b5, out_b);

    pod_trunk<<<N_COLS, 256, 0, stream>>>(coords, Wc, bc, Wm, bm, out_t, out_bias);

    const int nBlocks = (B_ROWS / 256) * (N_COLS / 128);  // 1024
    pod_gemm_wmma<<<nBlocks, dim3(32, 8), 0, stream>>>(out_b, out_t, out_bias, out);
}
